// SwinTransformerBlock_12807592477289
// MI455X (gfx1250) — matrix operand order repaired and checked by reading, not compiled, not run on hardware
//
#include <hip/hip_runtime.h>
#include <hip/hip_bf16.h>
#include <math.h>

// ---------------------------------------------------------------------------
// Swin Transformer block (B=32, H=W=56, C=384, heads=12, hd=32, WS=7, SS=3)
// f16 WMMA (v_wmma_f32_16x16x32_f16) for all matmuls, f32 accumulate,
// f32 residual stream. GEMMs use 32x64 wave tiles: 2 A-tiles share each
// B-tile (8 WMMA per 6 b128-load-pairs per K-chunk).
// ---------------------------------------------------------------------------

typedef __attribute__((ext_vector_type(16))) _Float16 v16h;
typedef __attribute__((ext_vector_type(8)))  _Float16 v8h;
typedef __attribute__((ext_vector_type(8)))  float    v8f;

union V16 { v16h v; v8h h[2]; };

#define DIMC   384
#define HEADS  12
#define HD     32
#define QKVC   1152
#define DFF    1536
#define NWIN   2048          // 32 batches * 64 windows
#define NPAD   64            // 49 tokens padded to 64
#define MROWS  (NWIN*NPAD)   // 131072
#define BATCH  32
#define LTOK   3136          // 56*56
#define TOK    (BATCH*LTOK)  // 100352
#define SCALE  0.17677669529663687f  // 1/sqrt(32)

__device__ inline v8f vzero8() {
  v8f z = {0.f,0.f,0.f,0.f,0.f,0.f,0.f,0.f};
  return z;
}

// A-matrix 16x32 f16 per ISA layout: lane L holds row m0+(L&15);
// elements 0..7 -> K = kbase..kbase+7, elements 8..15 -> K = kbase+16..+23,
// kbase = k0 + 8*(L>>4). Two contiguous 16B loads.
__device__ inline v16h loadA16x32(const _Float16* p, int ld, int m0, int k0, int lane) {
  const int row = m0 + (lane & 15);
  const int kb  = k0 + ((lane >> 4) << 3);
  const _Float16* r = p + (size_t)row * ld + kb;
  V16 a;
  a.h[0] = *(const v8h*)(r);
  a.h[1] = *(const v8h*)(r + 16);
  return a.v;
}

// B-matrix 32x16 f16 per ISA layout, sourced from a transposed array BT[N][K]:
// lane L holds col n0+(L&15); elements 0..7 -> K = k0 + 8*(L>>4) + e,
// elements 8..15 -> K = k0 + 8*(L>>4) + 16 + (e-8) (two 16B loads).
__device__ inline v16h loadB32x16T(const _Float16* pT, int ld, int k0, int n0, int lane) {
  const _Float16* r = pT + (size_t)(n0 + (lane & 15)) * ld + k0 + ((lane >> 4) << 3);
  V16 b;
  b.h[0] = *(const v8h*)(r);
  b.h[1] = *(const v8h*)(r + 16);
  return b.v;
}

// One wave computes a 32x64 f32 tile: acc[2][4] C-tiles, K-loop chunks of 32.
// Two A row-tiles share each B tile -> 8 WMMA per 6 load-pairs.
__device__ inline void gemm_core32(const _Float16* A, int lda,
                                   const _Float16* BT, int ldb,
                                   int K, int m0, int n0, int lane, v8f acc[2][4]) {
  for (int k = 0; k < K; k += 32) {
    v16h a0 = loadA16x32(A, lda, m0,      k, lane);
    v16h a1 = loadA16x32(A, lda, m0 + 16, k, lane);
#pragma unroll
    for (int t = 0; t < 4; ++t) {
      v16h b = loadB32x16T(BT, ldb, k, n0 + 16 * t, lane);
      acc[0][t] = __builtin_amdgcn_wmma_f32_16x16x32_f16(false, a0, false, b,
                                                         (short)0, acc[0][t], false, false);
      acc[1][t] = __builtin_amdgcn_wmma_f32_16x16x32_f16(false, a1, false, b,
                                                         (short)0, acc[1][t], false, false);
    }
  }
}

// winrow (0..131071) -> (batch, token index) after undoing shift+window.
// Returns -1 for pad rows (i >= 49).
__device__ inline int winrow_to_token(int row, int* bout) {
  int win = row >> 6, i = row & 63;
  if (i >= 49) return -1;
  int w  = win & 63;
  int iy = i / 7, ix = i - iy * 7;
  int y  = ((w >> 3) * 7) + iy;
  int x  = ((w & 7)  * 7) + ix;
  int yo = y + 3; if (yo >= 56) yo -= 56;
  int xo = x + 3; if (xo >= 56) xo -= 56;
  *bout = win >> 6;
  return yo * 56 + xo;
}

// region label for the shifted-window attention mask
__device__ inline int regionLabel(int edgeH, int edgeW, int p) {
  int py = p / 7, px = p - py * 7;
  int hr = edgeH ? (py < 4 ? 1 : 2) : 0;
  int wr = edgeW ? (px < 4 ? 1 : 2) : 0;
  return hr * 3 + wr;
}

// ---------------------------------------------------------------------------
// K-pre: weight convert (f32 -> f16, transposed to [N][K])
// ---------------------------------------------------------------------------
__global__ void convT_kernel(const float* __restrict__ w, _Float16* __restrict__ wT,
                             int K, int N) {
  int idx = blockIdx.x * 256 + threadIdx.x;
  if (idx >= K * N) return;
  int n = idx / K, k = idx - n * K;
  wT[idx] = (_Float16)w[(size_t)k * N + n];
}

// ---------------------------------------------------------------------------
// K-pre: combined relative-position-bias + shift mask table
// bm[type(4)][head(12)][i(64)][j(64)], j>=49 -> -1e5 (pad columns)
// ---------------------------------------------------------------------------
__global__ void build_bm_kernel(const int* __restrict__ rel_idx,
                                const float* __restrict__ rpb,
                                float* __restrict__ bm) {
  int idx = blockIdx.x * 256 + threadIdx.x;
  if (idx >= 4 * HEADS * 64 * 64) return;
  int t = idx / (HEADS * 4096);
  int h = (idx / 4096) % HEADS;
  int i = (idx >> 6) & 63;
  int j = idx & 63;
  float v = 0.f;
  if (j >= 49) {
    v = -1.0e5f;
  } else if (i < 49) {
    v = rpb[rel_idx[i * 49 + j] * HEADS + h];
    int eh = (t >> 1) & 1, ew = t & 1;
    if (regionLabel(eh, ew, i) != regionLabel(eh, ew, j)) v += -100.f;
  }
  bm[idx] = v;
}

// ---------------------------------------------------------------------------
// K0: LN1 + cyclic shift + window partition -> f16 winX[131072][384]
// One wave per window-row; pad rows (i>=49) are zeroed.
// ---------------------------------------------------------------------------
__global__ void ln1_window_kernel(const float* __restrict__ x,
                                  const float* __restrict__ g,
                                  const float* __restrict__ bt,
                                  _Float16* __restrict__ winX) {
  int wave = (blockIdx.x * blockDim.x + threadIdx.x) >> 5;
  int lane = threadIdx.x & 31;
  if (wave >= MROWS) return;
  _Float16* out = winX + (size_t)wave * DIMC;
  int b;
  int l = winrow_to_token(wave, &b);
  if (l < 0) {
#pragma unroll
    for (int t = 0; t < 12; ++t) out[t * 32 + lane] = (_Float16)0.f;
    return;
  }
  const float* xr = x + ((size_t)b * LTOK + l) * DIMC;
  float v[12];
  float s = 0.f;
#pragma unroll
  for (int t = 0; t < 12; ++t) { v[t] = xr[t * 32 + lane]; s += v[t]; }
#pragma unroll
  for (int off = 16; off >= 1; off >>= 1) s += __shfl_xor(s, off);
  float mu = s * (1.f / 384.f);
  float vs = 0.f;
#pragma unroll
  for (int t = 0; t < 12; ++t) { float d = v[t] - mu; vs += d * d; }
#pragma unroll
  for (int off = 16; off >= 1; off >>= 1) vs += __shfl_xor(vs, off);
  float rs = rsqrtf(vs * (1.f / 384.f) + 1e-5f);
#pragma unroll
  for (int t = 0; t < 12; ++t) {
    int c = t * 32 + lane;
    out[c] = (_Float16)((v[t] - mu) * rs * g[c] + bt[c]);
  }
}

// ---------------------------------------------------------------------------
// K1: QKV GEMM  (131072 x 384) @ (384 x 1152) -> f16 qkv
// ---------------------------------------------------------------------------
__global__ void gemm_qkv_kernel(const _Float16* __restrict__ A,
                                const _Float16* __restrict__ BT,
                                _Float16* __restrict__ C) {
  int wave = (blockIdx.x * blockDim.x + threadIdx.x) >> 5;
  int lane = threadIdx.x & 31;
  const int NG = QKVC / 64;  // 18
  int mt = wave / NG, ng = wave - mt * NG;
  if (mt >= MROWS / 32) return;
  int m0 = mt * 32, n0 = ng * 64;
  v8f acc[2][4] = {{vzero8(), vzero8(), vzero8(), vzero8()},
                   {vzero8(), vzero8(), vzero8(), vzero8()}};
  gemm_core32(A, DIMC, BT, DIMC, DIMC, m0, n0, lane, acc);
  int half = lane >> 4, jn = lane & 15;
#pragma unroll
  for (int rt = 0; rt < 2; ++rt)
#pragma unroll
    for (int t = 0; t < 4; ++t)
#pragma unroll
      for (int r = 0; r < 8; ++r)
        C[(size_t)(m0 + rt * 16 + r + 8 * half) * QKVC + n0 + t * 16 + jn] =
            (_Float16)acc[rt][t][r];
}

// ---------------------------------------------------------------------------
// K2: windowed attention. One block (4 waves) per (window, head);
// wave `wid` owns query-row tile ti=wid (16 rows). V staged transposed in LDS.
// ---------------------------------------------------------------------------
__global__ __launch_bounds__(128) void attn_kernel(const _Float16* __restrict__ qkv,
                                                   const float* __restrict__ bm,
                                                   _Float16* __restrict__ attnO) {
  __shared__ _Float16 sV[32 * 72];       // V transposed: sV[d*72 + j]
  __shared__ _Float16 sP[4][16 * 72];    // per-wave probability rows

  int tid = threadIdx.x;
  int wid = tid >> 5, lane = tid & 31;
  int half = lane >> 4, ln = lane & 15;
  int task = blockIdx.x;                 // 0 .. 24575
  int win = task / HEADS, h = task - win * HEADS;

  const _Float16* base = qkv + (size_t)win * NPAD * QKVC;
  const _Float16* qb = base + h * HD;
  const _Float16* kb = base + DIMC + h * HD;
  const _Float16* vb = base + 2 * DIMC + h * HD;

  // cooperative stage of V (64 tokens x 32 dims), transposed
  for (int t = tid; t < NPAD * HD; t += 128) {
    int j = t >> 5, d = t & 31;
    sV[d * 72 + j] = vb[(size_t)j * QKVC + d];
  }
  __syncthreads();

  int ti = wid;
  // S = Q K^T : K dim = 32 (one WMMA per 16x16 tile)
  v16h aq = loadA16x32(qb, QKVC, ti * 16, 0, lane);
  v8f s[4];
#pragma unroll
  for (int tj = 0; tj < 4; ++tj) {
    v16h bk = loadB32x16T(kb, QKVC, 0, tj * 16, lane);
    s[tj] = __builtin_amdgcn_wmma_f32_16x16x32_f16(false, aq, false, bk,
                                                   (short)0, vzero8(), false, false);
  }

  // scale + rel-pos bias + shift mask (combined table)
  int w = win & 63;
  int mt = (((w >> 3) == 7) ? 2 : 0) | (((w & 7) == 7) ? 1 : 0);
  const float* bmp = bm + ((size_t)mt * HEADS + h) * 4096;
#pragma unroll
  for (int tj = 0; tj < 4; ++tj)
#pragma unroll
    for (int r = 0; r < 8; ++r) {
      int i = ti * 16 + r + 8 * half;
      int j = tj * 16 + ln;
      s[tj][r] = s[tj][r] * SCALE + bmp[i * 64 + j];
    }

  // softmax: each row lives in one 16-lane half (4 regs per lane)
#pragma unroll
  for (int r = 0; r < 8; ++r) {
    float m = fmaxf(fmaxf(s[0][r], s[1][r]), fmaxf(s[2][r], s[3][r]));
#pragma unroll
    for (int off = 8; off >= 1; off >>= 1) m = fmaxf(m, __shfl_xor(m, off));
    float sum = 0.f;
#pragma unroll
    for (int tj = 0; tj < 4; ++tj) {
      float p = __expf(s[tj][r] - m);
      s[tj][r] = p;
      sum += p;
    }
#pragma unroll
    for (int off = 8; off >= 1; off >>= 1) sum += __shfl_xor(sum, off);
    float rinv = 1.f / sum;
    int ir = r + 8 * half;   // row within this wave's 16-row tile
#pragma unroll
    for (int tj = 0; tj < 4; ++tj)
      sP[wid][ir * 72 + tj * 16 + ln] = (_Float16)(s[tj][r] * rinv);
  }
  // per-wave LDS: same-wave DS ordering suffices, no block barrier needed

  // O = P V : K dim = 64 -> two chained WMMAs per output tile
  v8f o[2] = {vzero8(), vzero8()};
#pragma unroll
  for (int kc = 0; kc < 2; ++kc) {
    v16h ap = loadA16x32(&sP[wid][0], 72, 0, kc * 32, lane);
#pragma unroll
    for (int td = 0; td < 2; ++td) {
      v16h bv = loadB32x16T(sV, 72, kc * 32, td * 16, lane);
      o[td] = __builtin_amdgcn_wmma_f32_16x16x32_f16(false, ap, false, bv,
                                                     (short)0, o[td], false, false);
    }
  }

#pragma unroll
  for (int td = 0; td < 2; ++td)
#pragma unroll
    for (int r = 0; r < 8; ++r) {
      int i = ti * 16 + r + 8 * half;
      int col = h * HD + td * 16 + ln;
      attnO[((size_t)win * NPAD + i) * DIMC + col] = (_Float16)o[td][r];
    }
}

// ---------------------------------------------------------------------------
// K3: proj GEMM + bias + un-window/un-shift scatter + residual -> f32 h
// ---------------------------------------------------------------------------
__global__ void gemm_proj_kernel(const _Float16* __restrict__ A,
                                 const _Float16* __restrict__ BT,
                                 const float* __restrict__ pb,
                                 const float* __restrict__ x,
                                 float* __restrict__ hout) {
  int wave = (blockIdx.x * blockDim.x + threadIdx.x) >> 5;
  int lane = threadIdx.x & 31;
  const int NG = DIMC / 64;  // 6
  int mt = wave / NG, ng = wave - mt * NG;
  if (mt >= MROWS / 32) return;
  int m0 = mt * 32, n0 = ng * 64;
  v8f acc[2][4] = {{vzero8(), vzero8(), vzero8(), vzero8()},
                   {vzero8(), vzero8(), vzero8(), vzero8()}};
  gemm_core32(A, DIMC, BT, DIMC, DIMC, m0, n0, lane, acc);
  int half = lane >> 4, jn = lane & 15;
#pragma unroll
  for (int rt = 0; rt < 2; ++rt)
#pragma unroll
    for (int r = 0; r < 8; ++r) {
      int row = m0 + rt * 16 + r + 8 * half;
      int b;
      int l = winrow_to_token(row, &b);
      if (l < 0) continue;
      size_t ro = ((size_t)b * LTOK + l) * DIMC;
#pragma unroll
      for (int t = 0; t < 4; ++t) {
        int col = n0 + t * 16 + jn;
        hout[ro + col] = x[ro + col] + acc[rt][t][r] + pb[col];
      }
    }
}

// ---------------------------------------------------------------------------
// K4: LN2 -> f16 (plain token rows)
// ---------------------------------------------------------------------------
__global__ void ln2_kernel(const float* __restrict__ h,
                           const float* __restrict__ g,
                           const float* __restrict__ bt,
                           _Float16* __restrict__ out16) {
  int wave = (blockIdx.x * blockDim.x + threadIdx.x) >> 5;
  int lane = threadIdx.x & 31;
  if (wave >= TOK) return;
  const float* xr = h + (size_t)wave * DIMC;
  _Float16* out = out16 + (size_t)wave * DIMC;
  float v[12];
  float s = 0.f;
#pragma unroll
  for (int t = 0; t < 12; ++t) { v[t] = xr[t * 32 + lane]; s += v[t]; }
#pragma unroll
  for (int off = 16; off >= 1; off >>= 1) s += __shfl_xor(s, off);
  float mu = s * (1.f / 384.f);
  float vs = 0.f;
#pragma unroll
  for (int t = 0; t < 12; ++t) { float d = v[t] - mu; vs += d * d; }
#pragma unroll
  for (int off = 16; off >= 1; off >>= 1) vs += __shfl_xor(vs, off);
  float rs = rsqrtf(vs * (1.f / 384.f) + 1e-5f);
#pragma unroll
  for (int t = 0; t < 12; ++t) {
    int c = t * 32 + lane;
    out[c] = (_Float16)((v[t] - mu) * rs * g[c] + bt[c]);
  }
}

// ---------------------------------------------------------------------------
// K5: FFN1 GEMM + bias + exact GELU -> f16 (100352 x 1536)
// ---------------------------------------------------------------------------
__global__ void gemm_ffn1_kernel(const _Float16* __restrict__ A,
                                 const _Float16* __restrict__ BT,
                                 const float* __restrict__ b1,
                                 _Float16* __restrict__ ffh) {
  int wave = (blockIdx.x * blockDim.x + threadIdx.x) >> 5;
  int lane = threadIdx.x & 31;
  const int NG = DFF / 64;  // 24
  int mt = wave / NG, ng = wave - mt * NG;
  if (mt >= TOK / 32) return;
  int m0 = mt * 32, n0 = ng * 64;
  v8f acc[2][4] = {{vzero8(), vzero8(), vzero8(), vzero8()},
                   {vzero8(), vzero8(), vzero8(), vzero8()}};
  gemm_core32(A, DIMC, BT, DIMC, DIMC, m0, n0, lane, acc);
  int half = lane >> 4, jn = lane & 15;
#pragma unroll
  for (int rt = 0; rt < 2; ++rt)
#pragma unroll
    for (int t = 0; t < 4; ++t)
#pragma unroll
      for (int r = 0; r < 8; ++r) {
        int row = m0 + rt * 16 + r + 8 * half;
        int col = n0 + t * 16 + jn;
        float u = acc[rt][t][r] + b1[col];
        float ge = 0.5f * u * (1.f + erff(u * 0.70710678118f));
        ffh[(size_t)row * DFF + col] = (_Float16)ge;
      }
}

// ---------------------------------------------------------------------------
// K6: FFN2 GEMM + bias + residual -> f32 d_out
// ---------------------------------------------------------------------------
__global__ void gemm_ffn2_kernel(const _Float16* __restrict__ A,
                                 const _Float16* __restrict__ BT,
                                 const float* __restrict__ b2,
                                 const float* __restrict__ h,
                                 float* __restrict__ out) {
  int wave = (blockIdx.x * blockDim.x + threadIdx.x) >> 5;
  int lane = threadIdx.x & 31;
  const int NG = DIMC / 64;  // 6
  int mt = wave / NG, ng = wave - mt * NG;
  if (mt >= TOK / 32) return;
  int m0 = mt * 32, n0 = ng * 64;
  v8f acc[2][4] = {{vzero8(), vzero8(), vzero8(), vzero8()},
                   {vzero8(), vzero8(), vzero8(), vzero8()}};
  gemm_core32(A, DFF, BT, DFF, DFF, m0, n0, lane, acc);
  int half = lane >> 4, jn = lane & 15;
#pragma unroll
  for (int rt = 0; rt < 2; ++rt)
#pragma unroll
    for (int t = 0; t < 4; ++t)
#pragma unroll
      for (int r = 0; r < 8; ++r) {
        int row = m0 + rt * 16 + r + 8 * half;
        int col = n0 + t * 16 + jn;
        size_t o = (size_t)row * DIMC + col;
        out[o] = h[o] + acc[rt][t][r] + b2[col];
      }
}

// ---------------------------------------------------------------------------
// Launch
// ---------------------------------------------------------------------------
extern "C" void kernel_launch(void* const* d_in, const int* in_sizes, int n_in,
                              void* d_out, int out_size, void* d_ws, size_t ws_size,
                              hipStream_t stream) {
  const float* x      = (const float*)d_in[0];
  const int*   relidx = (const int*)  d_in[3];
  const float* rpb    = (const float*)d_in[4];
  const float* qkv_w  = (const float*)d_in[5];
  const float* proj_w = (const float*)d_in[6];
  const float* proj_b = (const float*)d_in[7];
  const float* n1g    = (const float*)d_in[8];
  const float* n1b    = (const float*)d_in[9];
  const float* n2g    = (const float*)d_in[10];
  const float* n2b    = (const float*)d_in[11];
  const float* ffw1   = (const float*)d_in[12];
  const float* ffb1   = (const float*)d_in[13];
  const float* ffw2   = (const float*)d_in[14];
  const float* ffb2   = (const float*)d_in[15];
  float* out = (float*)d_out;

  char* wsb = (char*)d_ws;
  size_t off = 0;
  auto take = [&](size_t bytes) -> void* {
    void* p = wsb + off;
    off += (bytes + 255) & ~(size_t)255;
    return p;
  };
  _Float16* qkvT  = (_Float16*)take((size_t)DIMC * QKVC * 2);
  _Float16* projT = (_Float16*)take((size_t)DIMC * DIMC * 2);
  _Float16* f1T   = (_Float16*)take((size_t)DIMC * DFF * 2);
  _Float16* f2T   = (_Float16*)take((size_t)DFF * DIMC * 2);
  float*    bm    = (float*)   take((size_t)4 * HEADS * 64 * 64 * 4);
  _Float16* bufA  = (_Float16*)take((size_t)MROWS * DIMC * 2);  // winX, later LN2 out
  _Float16* bufB  = (_Float16*)take((size_t)TOK * DFF * 2);     // qkv buf, later FFN hidden
  _Float16* attnO = (_Float16*)take((size_t)MROWS * DIMC * 2);
  float*    hbuf  = (float*)   take((size_t)TOK * DIMC * 4);
  if (off > ws_size) return;  // workspace too small: bail silently

  _Float16* winX   = bufA;
  _Float16* qkvbuf = bufB;   // 131072*1152 f16 fits inside TOK*DFF
  _Float16* g16    = bufA;
  _Float16* ffh    = bufB;

  // weight conversion (transposed f16)
  convT_kernel<<<(DIMC * QKVC + 255) / 256, 256, 0, stream>>>(qkv_w, qkvT, DIMC, QKVC);
  convT_kernel<<<(DIMC * DIMC + 255) / 256, 256, 0, stream>>>(proj_w, projT, DIMC, DIMC);
  convT_kernel<<<(DIMC * DFF + 255) / 256, 256, 0, stream>>>(ffw1, f1T, DIMC, DFF);
  convT_kernel<<<(DFF * DIMC + 255) / 256, 256, 0, stream>>>(ffw2, f2T, DFF, DIMC);

  // bias+mask table
  build_bm_kernel<<<(4 * HEADS * 4096 + 255) / 256, 256, 0, stream>>>(relidx, rpb, bm);

  // LN1 + shift + window
  ln1_window_kernel<<<MROWS / 8, 256, 0, stream>>>(x, n1g, n1b, winX);

  // QKV: (131072/32) * (1152/64) waves, 8 waves per block
  gemm_qkv_kernel<<<(MROWS / 32) * (QKVC / 64) / 8, 256, 0, stream>>>(winX, qkvT, qkvbuf);

  // attention
  attn_kernel<<<NWIN * HEADS, 128, 0, stream>>>(qkvbuf, bm, attnO);

  // proj + scatter + residual
  gemm_proj_kernel<<<(MROWS / 32) * (DIMC / 64) / 8, 256, 0, stream>>>(attnO, projT,
                                                                       proj_b, x, hbuf);
  // LN2
  ln2_kernel<<<TOK / 8, 256, 0, stream>>>(hbuf, n2g, n2b, g16);

  // FFN
  gemm_ffn1_kernel<<<(TOK / 32) * (DFF / 64) / 8, 256, 0, stream>>>(g16, f1T, ffb1, ffh);
  gemm_ffn2_kernel<<<(TOK / 32) * (DIMC / 64) / 8, 256, 0, stream>>>(ffh, f2T, ffb2,
                                                                     hbuf, out);
  (void)in_sizes; (void)n_in; (void)out_size;
}
